// SB_Layer_11982958756754
// MI455X (gfx1250) — matrix operand order repaired and checked by reading, not compiled, not run on hardware
//
#include <hip/hip_runtime.h>
#include <stdint.h>

#define EPSF   1e-8f
#define TEMPB  0.67f
#define TEMPC  0.67f
#define DATAN  60000.0f

#define Bsz 4096
#define Dd  4096
#define Kk  2048
#define Nn  4096   /* K*U */

typedef __attribute__((ext_vector_type(16))) __bf16 bf16x16;
typedef __attribute__((ext_vector_type(8)))  float  floatx8;

union Frag16 { bf16x16 v; uint4 q[2]; };

// ---------------- math helpers (match jax reference numerics) ----------------
__device__ __forceinline__ float softplusf(float x) {
  return fmaxf(x, 0.f) + log1pf(expf(-fabsf(x)));
}
__device__ __forceinline__ float sigmoidf(float x) {
  return 1.f / (1.f + expf(-x));
}
__device__ __forceinline__ float bc_sample(float a, float u) {
  float L = logf(u + EPSF) - logf(1.f - u + EPSF);
  float x = sigmoidf((L + logf(a + EPSF)) / TEMPB);
  return fminf(fmaxf(x, 1e-4f), 1.f - 1e-4f);
}
__device__ __forceinline__ float bc_lpdf(float x, float a) {
  return logf(TEMPB) + logf(a + EPSF)
       - (TEMPB + 1.f) * (logf(x + EPSF) + logf(1.f - x + EPSF))
       - 2.f * logf(a * powf(x, -TEMPB) + powf(1.f - x, -TEMPB) + EPSF);
}

// deterministic 256-thread block reduction (fixed-order LDS tree)
__device__ float blockReduce256(float v) {
  __shared__ float red[256];
  __syncthreads();
  red[threadIdx.x] = v;
  __syncthreads();
  for (int s = 128; s > 0; s >>= 1) {
    if (threadIdx.x < s) red[threadIdx.x] += red[threadIdx.x + s];
    __syncthreads();
  }
  return red[0];
}

// ---------------- kernel 1: inputs fp32 -> bf16 -----------------------------
__global__ __launch_bounds__(256) void k_cast(const float* __restrict__ in,
                                              __bf16* __restrict__ out) {
  int i = blockIdx.x * 256 + threadIdx.x;          // 4 elems each, n4 = 4194304
  float4 v = ((const float4*)in)[i];
  union { __bf16 h[4]; uint2 u; } o;
  o.h[0] = (__bf16)v.x; o.h[1] = (__bf16)v.y;
  o.h[2] = (__bf16)v.z; o.h[3] = (__bf16)v.w;
  *(uint2*)(out + 4 * i) = o.u;
}

// ---------------- kernel 2: chunk products of q_u + kl_sticks ---------------
__global__ __launch_bounds__(256) void k_sticks1(const float* __restrict__ conc1,
                                                 const float* __restrict__ conc0,
                                                 const float* __restrict__ u_kum,
                                                 float* __restrict__ chunkprod,
                                                 float* __restrict__ p_sticks) {
  int tid = blockIdx.x * 256 + threadIdx.x;   // 65536 = 32 chunks x 2048 cols
  int k = tid & (Kk - 1);
  int c = tid >> 11;
  float c1 = softplusf(conc1[k]);
  float c0 = softplusf(conc0[k]);
  float ic1 = 1.f / c1, ic0 = 1.f / c0;
  float lpp = logf(1.f + EPSF);               // kuma log-pdf at a=b=1
  float prod = 1.f, kls = 0.f;
  int base = c * 128;
  for (int i = 0; i < 128; ++i) {
    float u = u_kum[(base + i) * Kk + k];
    float x = powf(1.f - powf(u, ic0), ic1);  // kumaraswamy_sample
    prod *= x;
    float qlp = logf(c1 * c0 + EPSF) + (c1 - 1.f) * logf(x + EPSF)
              + (c0 - 1.f) * logf(1.f - powf(x, c1) + EPSF);
    kls += qlp - lpp;
  }
  chunkprod[c * Kk + k] = prod;
  float bs = blockReduce256(kls);
  if (threadIdx.x == 0) p_sticks[blockIdx.x] = bs;
}

// ---------------- kernel 3: exclusive scan over the 32 chunks ---------------
__global__ __launch_bounds__(256) void k_scan(const float* __restrict__ chunkprod,
                                              float* __restrict__ chunkpref) {
  int k = blockIdx.x * 256 + threadIdx.x;     // 2048 threads
  float run = 1.f;
  for (int c = 0; c < 32; ++c) {
    chunkpref[c * Kk + k] = run;
    run *= chunkprod[c * Kk + k];
  }
}

// ---------------- kernel 4: replay -> pi, z, kl_z ---------------------------
__global__ __launch_bounds__(256) void k_sticks2(const float* __restrict__ conc1,
                                                 const float* __restrict__ conc0,
                                                 const float* __restrict__ u_kum,
                                                 const float* __restrict__ t_pi,
                                                 const float* __restrict__ u_bern,
                                                 const float* __restrict__ chunkpref,
                                                 float* __restrict__ p_z) {
  int tid = blockIdx.x * 256 + threadIdx.x;
  int k = tid & (Kk - 1);
  int c = tid >> 11;
  float c1 = softplusf(conc1[k]);
  float c0 = softplusf(conc0[k]);
  float ic1 = 1.f / c1, ic0 = 1.f / c0;
  float run = chunkpref[c * Kk + k];
  float klz = 0.f;
  int base = c * 128;
  for (int i = 0; i < 128; ++i) {
    int d = base + i;
    float u = u_kum[d * Kk + k];
    float x = powf(1.f - powf(u, ic0), ic1);
    run *= x;                                  // run == pi[d,k]
    float a = sigmoidf(t_pi[d * Kk + k]);
    float z = bc_sample(a, u_bern[d * Kk + k]);
    klz += bc_lpdf(z, a) - bc_lpdf(z, run);
  }
  float bs = blockReduce256(klz);
  if (threadIdx.x == 0) p_z[blockIdx.x] = bs;
}

// ------- kernel 5: re = z * (mW + eps*softplus(sW)) -> bf16 transposed ------
__global__ __launch_bounds__(256) void k_re(const float* __restrict__ mW,
                                            const float* __restrict__ sW,
                                            const float* __restrict__ eps,
                                            const float* __restrict__ t_pi,
                                            const float* __restrict__ u_bern,
                                            __bf16* __restrict__ reBt,
                                            float* __restrict__ p_w) {
  __shared__ float tile[64][65];
  int j0 = blockIdx.x * 64;                   // N dim (columns of re)
  int d0 = blockIdx.y * 64;                   // D dim
  int t  = threadIdx.x;
  int zj0 = j0 & (Kk - 1);                    // j % K (tile(z,(1,U)))
  float klw = 0.f;
  for (int i = 0; i < 16; ++i) {
    int idx = i * 256 + t;
    int dl = idx >> 6, jl = idx & 63;
    int g  = (d0 + dl) * Nn + (j0 + jl);
    float mw = mW[g], sw = sW[g], ep = eps[g];
    float sp = softplusf(sw);
    float w  = mw + ep * sp;
    int gz = (d0 + dl) * Kk + (zj0 + jl);
    float a = sigmoidf(t_pi[gz]);
    float z = bc_sample(a, u_bern[gz]);
    tile[dl][jl] = z * w;
    klw += 2.f * sp - mw * mw - sp * sp + 1.f;
  }
  __syncthreads();
  for (int i = 0; i < 16; ++i) {
    int idx = i * 256 + t;
    int jl = idx >> 6, dl = idx & 63;         // coalesced along d
    reBt[(j0 + jl) * Dd + (d0 + dl)] = (__bf16)tile[dl][jl];
  }
  float bs = blockReduce256(klw);
  if (t == 0) p_w[blockIdx.y * 64 + blockIdx.x] = bs;
}

// ---------------- kernel 6: lam = A(bf16) @ reBt^T via WMMA -----------------
#define LDSTR 40   // 32 K + 8 pad bf16 -> 80B rows, conflict-free b128 reads
__global__ __launch_bounds__(256) void k_gemm(const __bf16* __restrict__ A,
                                              const __bf16* __restrict__ Bt,
                                              float* __restrict__ C) {
  __shared__ __align__(16) __bf16 As[128 * LDSTR];
  __shared__ __align__(16) __bf16 Bs[128 * LDSTR];
  const int t    = threadIdx.x;
  const int m0   = blockIdx.y * 128;
  const int n0   = blockIdx.x * 128;
  const int lane = t & 31;
  const int wave = t >> 5;
  const int wm   = wave >> 1;                 // 4 row-waves
  const int wn   = wave & 1;                  // 2 col-waves

  // staging: 512 x 16B loads per tile; 2 per thread
  const int r0 = t >> 2,          q0 = t & 3;
  const int r1 = (t + 256) >> 2,  q1 = (t + 256) & 3;

  uint4 a0 = *(const uint4*)(A  + (m0 + r0) * Dd + q0 * 8);
  uint4 a1 = *(const uint4*)(A  + (m0 + r1) * Dd + q1 * 8);
  uint4 b0 = *(const uint4*)(Bt + (n0 + r0) * Dd + q0 * 8);
  uint4 b1 = *(const uint4*)(Bt + (n0 + r1) * Dd + q1 * 8);

  floatx8 acc[2][4];
#pragma unroll
  for (int mi = 0; mi < 2; ++mi)
#pragma unroll
    for (int ni = 0; ni < 4; ++ni)
      acc[mi][ni] = (floatx8){0.f,0.f,0.f,0.f,0.f,0.f,0.f,0.f};

  const int laneRow = lane & 15;
  const int hihalf  = lane >> 4;
  const int acb = hihalf * 8;    // A frag: K halves {0..7,16..23} / {8..15,24..31}
  const int bcb = hihalf * 8;    // B frag: K halves {0..7,16..23} / {8..15,24..31}

  for (int kk = 0; kk < Dd; kk += 32) {
    __syncthreads();
    *(uint4*)(As + r0 * LDSTR + q0 * 8) = a0;
    *(uint4*)(As + r1 * LDSTR + q1 * 8) = a1;
    *(uint4*)(Bs + r0 * LDSTR + q0 * 8) = b0;
    *(uint4*)(Bs + r1 * LDSTR + q1 * 8) = b1;
    __syncthreads();
    if (kk + 32 < Dd) {                       // prefetch next tile over compute
      int kn = kk + 32;
      a0 = *(const uint4*)(A  + (m0 + r0) * Dd + kn + q0 * 8);
      a1 = *(const uint4*)(A  + (m0 + r1) * Dd + kn + q1 * 8);
      b0 = *(const uint4*)(Bt + (n0 + r0) * Dd + kn + q0 * 8);
      b1 = *(const uint4*)(Bt + (n0 + r1) * Dd + kn + q1 * 8);
    }
    Frag16 af[2], bfg[4];
#pragma unroll
    for (int mi = 0; mi < 2; ++mi) {
      const __bf16* p = As + (wm * 32 + mi * 16 + laneRow) * LDSTR;
      af[mi].q[0] = *(const uint4*)(p + acb);
      af[mi].q[1] = *(const uint4*)(p + acb + 16);
    }
#pragma unroll
    for (int ni = 0; ni < 4; ++ni) {
      const __bf16* p = Bs + (wn * 64 + ni * 16 + laneRow) * LDSTR;
      bfg[ni].q[0] = *(const uint4*)(p + bcb);
      bfg[ni].q[1] = *(const uint4*)(p + bcb + 16);
    }
#pragma unroll
    for (int mi = 0; mi < 2; ++mi)
#pragma unroll
      for (int ni = 0; ni < 4; ++ni)
        acc[mi][ni] = __builtin_amdgcn_wmma_f32_16x16x32_bf16(
            false, af[mi].v, false, bfg[ni].v, (short)0, acc[mi][ni],
            false, false);
  }

  const int mBase = m0 + wm * 32 + hihalf * 8;
  const int nBase = n0 + wn * 64 + laneRow;
#pragma unroll
  for (int mi = 0; mi < 2; ++mi)
#pragma unroll
    for (int ni = 0; ni < 4; ++ni)
#pragma unroll
      for (int r = 0; r < 8; ++r)
        C[(mBase + mi * 16 + r) * Nn + nBase + ni * 16] = acc[mi][ni][r];
}

// ---- kernel 7: in-place epilogue (bias, softmax, concrete, out, kl_xi) -----
__global__ __launch_bounds__(256) void k_epi(float* __restrict__ lamout,
                                             const float* __restrict__ u_cat,
                                             const float* __restrict__ biases,
                                             float* __restrict__ p_xi) {
  const int total  = Bsz * Kk;                // 8388608 pairs
  const int stride = 2048 * 256;
  int tid = blockIdx.x * 256 + threadIdx.x;
  float kl = 0.f;
  const float lpr = logf(0.5f + EPSF);
  for (int e = tid; e < total; e += stride) {
    int b = e >> 11, k = e & (Kk - 1);
    float2 lam = *(float2*)(lamout + b * Nn + 2 * k);
    float2 bb  = *(const float2*)(biases + 2 * k);
    float l0 = lam.x + bb.x, l1 = lam.y + bb.y;
    float m = fmaxf(l0, l1);
    float e0 = expf(l0 - m), e1 = expf(l1 - m);
    float s = e0 + e1;
    float p0 = e0 / s + 1e-4f, p1 = e1 / s + 1e-4f;
    float ps = p0 + p1; p0 /= ps; p1 /= ps;
    float2 u = *(const float2*)(u_cat + 2 * e);
    float g0 = -logf(-logf(u.x + EPSF) + EPSF);
    float g1 = -logf(-logf(u.y + EPSF) + EPSF);
    float q0 = (logf(p0 + EPSF) + g0) / TEMPC;
    float q1 = (logf(p1 + EPSF) + g1) / TEMPC;
    float mq = fmaxf(q0, q1);
    float x0 = expf(q0 - mq), x1 = expf(q1 - mq);
    float xs = x0 + x1; x0 /= xs; x1 /= xs;
    float2 o; o.x = l0 * x0; o.y = l1 * x1;
    *(float2*)(lamout + b * Nn + 2 * k) = o;
    kl += p0 * (logf(p0 + EPSF) - lpr) + p1 * (logf(p1 + EPSF) - lpr);
  }
  float bs = blockReduce256(kl);
  if (threadIdx.x == 0) p_xi[blockIdx.x] = bs;
}

// ---------------- kernel 8: deterministic final loss ------------------------
__global__ __launch_bounds__(256) void k_loss(const float* __restrict__ p_sticks,
                                              const float* __restrict__ p_z,
                                              const float* __restrict__ p_w,
                                              const float* __restrict__ p_xi,
                                              float* __restrict__ out_loss) {
  float a = 0.f;
  for (int i = threadIdx.x; i < 256; i += 256) a += p_sticks[i];
  float s_sticks = blockReduce256(a);
  float b = 0.f;
  for (int i = threadIdx.x; i < 256; i += 256) b += p_z[i];
  float s_z = blockReduce256(b);
  float c = 0.f;
  for (int i = threadIdx.x; i < 4096; i += 256) c += p_w[i];
  float s_w = blockReduce256(c);
  float d = 0.f;
  for (int i = threadIdx.x; i < 2048; i += 256) d += p_xi[i];
  float s_xi = blockReduce256(d);
  if (threadIdx.x == 0) {
    float klw = -0.5f * (s_w / (float)(Dd * (long)Nn));
    float loss = (s_sticks + s_z + klw) / DATAN
               + (s_xi / (float)Bsz) / DATAN;
    out_loss[0] = loss;
  }
}

// ---------------------------------------------------------------------------
extern "C" void kernel_launch(void* const* d_in, const int* in_sizes, int n_in,
                              void* d_out, int out_size, void* d_ws, size_t ws_size,
                              hipStream_t stream) {
  const float* inputs = (const float*)d_in[0];
  const float* mW     = (const float*)d_in[1];
  const float* sW     = (const float*)d_in[2];
  const float* conc1  = (const float*)d_in[3];
  const float* conc0  = (const float*)d_in[4];
  const float* t_pi   = (const float*)d_in[5];
  const float* biases = (const float*)d_in[6];
  const float* eps    = (const float*)d_in[7];
  const float* u_kum  = (const float*)d_in[8];
  const float* u_bern = (const float*)d_in[9];
  const float* u_cat  = (const float*)d_in[10];
  float* out = (float*)d_out;

  char* ws = (char*)d_ws;
  __bf16* Abf      = (__bf16*)(ws);                         // 33,554,432 B
  __bf16* reBt     = (__bf16*)(ws + 33554432);              // 33,554,432 B
  float*  chunkprod = (float*)(ws + 67108864);              //    262,144 B
  float*  chunkpref = (float*)(ws + 67371008);              //    262,144 B
  float*  p_sticks  = (float*)(ws + 67633152);              //      1,024 B
  float*  p_z       = (float*)(ws + 67634176);              //      1,024 B
  float*  p_w       = (float*)(ws + 67635200);              //     16,384 B
  float*  p_xi      = (float*)(ws + 67651584);              //      8,192 B

  k_cast<<<(Bsz * Dd) / (4 * 256), 256, 0, stream>>>(inputs, Abf);
  k_sticks1<<<(32 * Kk) / 256, 256, 0, stream>>>(conc1, conc0, u_kum,
                                                 chunkprod, p_sticks);
  k_scan<<<Kk / 256, 256, 0, stream>>>(chunkprod, chunkpref);
  k_sticks2<<<(32 * Kk) / 256, 256, 0, stream>>>(conc1, conc0, u_kum, t_pi,
                                                 u_bern, chunkpref, p_z);
  k_re<<<dim3(Nn / 64, Dd / 64), 256, 0, stream>>>(mW, sW, eps, t_pi, u_bern,
                                                   reBt, p_w);
  k_gemm<<<dim3(Nn / 128, Bsz / 128), 256, 0, stream>>>(Abf, reBt, out);
  k_epi<<<2048, 256, 0, stream>>>(out, u_cat, biases, p_xi);
  k_loss<<<1, 256, 0, stream>>>(p_sticks, p_z, p_w, p_xi, out + (size_t)Bsz * Nn);
}
